// AdaptiveFFN_60713657697114
// MI455X (gfx1250) — matrix operand order repaired and checked by reading, not compiled, not run on hardware
//
#include <hip/hip_runtime.h>
#include <stdint.h>

#define H_DIM 1024
#define F_DIM 4096
#define NTOK  8192
#define LN_EPS 1e-5f

#define BM 256
#define BN 128
#define BK 32

typedef __attribute__((ext_vector_type(16))) __bf16 v16bf;
typedef __attribute__((ext_vector_type(8)))  float v8f;

union Frag {
    uint4 q[2];   // 32 bytes = 16 bf16
    v16bf v;
};

static __device__ __forceinline__ unsigned short f2bf(float f) {
    unsigned int u = __float_as_uint(f);
    unsigned int r = u + 0x7FFFu + ((u >> 16) & 1u);   // round-to-nearest-even
    return (unsigned short)(r >> 16);
}
static __device__ __forceinline__ float bf2f(unsigned short s) {
    return __uint_as_float(((unsigned int)s) << 16);
}
static __device__ __forceinline__ float gelu_exact(float x) {
    return 0.5f * x * (1.0f + erff(x * 0.70710678118654752f));
}

// ---------------------------------------------------------------------------
// Weight convert + transpose:  W[K x F] fp32 (row-major)  ->  Wt[F x K] bf16
// ---------------------------------------------------------------------------
__global__ __launch_bounds__(256) void wt_transpose_kernel(
    const float* __restrict__ W, unsigned short* __restrict__ Wt, int K, int F)
{
    __shared__ float tile[32][33];
    int f0 = blockIdx.x * 32, k0 = blockIdx.y * 32;
    int tx = threadIdx.x, ty = threadIdx.y;          // block = (32, 8)
#pragma unroll
    for (int i = 0; i < 4; ++i) {
        int kl = ty + i * 8;
        tile[kl][tx] = W[(size_t)(k0 + kl) * F + (f0 + tx)];
    }
    __syncthreads();
#pragma unroll
    for (int i = 0; i < 4; ++i) {
        int fl = ty + i * 8;
        Wt[(size_t)(f0 + fl) * K + (k0 + tx)] = f2bf(tile[tx][fl]);
    }
}

// ---------------------------------------------------------------------------
// LayerNorm over H=1024: writes fp32 and bf16 copies of x_norm
// ---------------------------------------------------------------------------
__global__ __launch_bounds__(256) void ln_x_kernel(
    const float* __restrict__ x, const float* __restrict__ g, const float* __restrict__ b,
    float* __restrict__ xn_f, unsigned short* __restrict__ xn_b)
{
    __shared__ float r1[256], r2[256];
    int row = blockIdx.x, tid = threadIdx.x;
    const float4* xr = (const float4*)(x + (size_t)row * H_DIM);
    float4 xv = xr[tid];
    float s  = xv.x + xv.y + xv.z + xv.w;
    float sq = xv.x * xv.x + xv.y * xv.y + xv.z * xv.z + xv.w * xv.w;
    r1[tid] = s; r2[tid] = sq;
    __syncthreads();
    for (int off = 128; off > 0; off >>= 1) {
        if (tid < off) { r1[tid] += r1[tid + off]; r2[tid] += r2[tid + off]; }
        __syncthreads();
    }
    float mean = r1[0] * (1.0f / H_DIM);
    float var  = r2[0] * (1.0f / H_DIM) - mean * mean;
    float rstd = rsqrtf(var + LN_EPS);
    float4 gv = ((const float4*)g)[tid];
    float4 bv = ((const float4*)b)[tid];
    float4 y;
    y.x = (xv.x - mean) * rstd * gv.x + bv.x;
    y.y = (xv.y - mean) * rstd * gv.y + bv.y;
    y.z = (xv.z - mean) * rstd * gv.z + bv.z;
    y.w = (xv.w - mean) * rstd * gv.w + bv.w;
    ((float4*)(xn_f + (size_t)row * H_DIM))[tid] = y;
    uint2 p;
    p.x = (unsigned)f2bf(y.x) | ((unsigned)f2bf(y.y) << 16);
    p.y = (unsigned)f2bf(y.z) | ((unsigned)f2bf(y.w) << 16);
    ((uint2*)(xn_b + (size_t)row * H_DIM))[tid] = p;
}

// ---------------------------------------------------------------------------
// LayerNorm over F=4096, bf16 in -> bf16 out
// ---------------------------------------------------------------------------
__global__ __launch_bounds__(256) void ln_h_kernel(
    const unsigned short* __restrict__ h, const float* __restrict__ g,
    const float* __restrict__ b, unsigned short* __restrict__ hn)
{
    __shared__ float r1[256], r2[256];
    int row = blockIdx.x, tid = threadIdx.x;
    const uint4* hr = (const uint4*)(h + (size_t)row * F_DIM);
    uint4 q0 = hr[tid * 2], q1 = hr[tid * 2 + 1];
    float v[16];
    const unsigned int* w0 = (const unsigned int*)&q0;
    const unsigned int* w1 = (const unsigned int*)&q1;
#pragma unroll
    for (int i = 0; i < 4; ++i) {
        v[2 * i]     = bf2f((unsigned short)(w0[i] & 0xFFFF));
        v[2 * i + 1] = bf2f((unsigned short)(w0[i] >> 16));
        v[8 + 2 * i]     = bf2f((unsigned short)(w1[i] & 0xFFFF));
        v[8 + 2 * i + 1] = bf2f((unsigned short)(w1[i] >> 16));
    }
    float s = 0.f, sq = 0.f;
#pragma unroll
    for (int i = 0; i < 16; ++i) { s += v[i]; sq += v[i] * v[i]; }
    r1[tid] = s; r2[tid] = sq;
    __syncthreads();
    for (int off = 128; off > 0; off >>= 1) {
        if (tid < off) { r1[tid] += r1[tid + off]; r2[tid] += r2[tid + off]; }
        __syncthreads();
    }
    float mean = r1[0] * (1.0f / F_DIM);
    float var  = r2[0] * (1.0f / F_DIM) - mean * mean;
    float rstd = rsqrtf(var + LN_EPS);
    int c = tid * 16;
    unsigned int out[8];
#pragma unroll
    for (int i = 0; i < 8; ++i) {
        float y0 = (v[2 * i]     - mean) * rstd * g[c + 2 * i]     + b[c + 2 * i];
        float y1 = (v[2 * i + 1] - mean) * rstd * g[c + 2 * i + 1] + b[c + 2 * i + 1];
        out[i] = (unsigned)f2bf(y0) | ((unsigned)f2bf(y1) << 16);
    }
    uint4* hw = (uint4*)(hn + (size_t)row * F_DIM);
    hw[tid * 2]     = make_uint4(out[0], out[1], out[2], out[3]);
    hw[tid * 2 + 1] = make_uint4(out[4], out[5], out[6], out[7]);
}

// ---------------------------------------------------------------------------
// WMMA bf16 GEMM:  C[M x N] = A[M x K] * Bt[N x K]^T + bias
//   mode 0: store fp32
//   mode 1: store bf16 of gelu(result)
// Block: 256 threads = 8 waves; tile 256x128, BK=32.
// Wave (wrow 0..3, wcol 0..1) owns a 64x64 sub-tile = 4x4 WMMA accumulators:
// 16 v_wmma per K-step against 16 ds_load_b128 (1.0 LDS loads per WMMA).
// ---------------------------------------------------------------------------
__global__ __launch_bounds__(256) void gemm_bf16_kernel(
    const unsigned short* __restrict__ A, const unsigned short* __restrict__ Bt,
    const float* __restrict__ bias, void* __restrict__ Cout,
    int M, int N, int K, int gelu_bf16_out)
{
    __shared__ __align__(16) unsigned short As[BM * BK];   // 16 KB
    __shared__ __align__(16) unsigned short Bs[BN * BK];   //  8 KB

    int tid  = threadIdx.x;
    int lane = tid & 31;
    int wave = tid >> 5;
    int wrow = wave >> 1;          // 0..3  (rows of 64)
    int wcol = wave & 1;           // 0..1  (cols of 64)
    int hi   = lane >> 4;          // K-half select per ISA layout
    int l15  = lane & 15;

    int row0 = blockIdx.y * BM;
    int col0 = blockIdx.x * BN;

    v8f acc[4][4];
#pragma unroll
    for (int m = 0; m < 4; ++m)
#pragma unroll
        for (int n = 0; n < 4; ++n)
#pragma unroll
            for (int r = 0; r < 8; ++r) acc[m][n][r] = 0.0f;

    // A tile: 256 rows x 32 cols = 512 chunks of 16 bf16; 2 chunks per thread.
    int ca0 = tid, ca1 = tid + 256;
    int ar0 = ca0 >> 1, ac0 = (ca0 & 1) * 16;
    int ar1 = ca1 >> 1, ac1 = (ca1 & 1) * 16;
    // B tile: 128 rows x 32 cols = 256 chunks; 1 chunk per thread.
    int br = tid >> 1, bc = (tid & 1) * 16;

    const unsigned short* Ag0 = A  + (size_t)(row0 + ar0) * K + ac0;
    const unsigned short* Ag1 = A  + (size_t)(row0 + ar1) * K + ac1;
    const unsigned short* Bg  = Bt + (size_t)(col0 + br ) * K + bc;
    unsigned short* AsW0 = &As[ar0 * BK + ac0];
    unsigned short* AsW1 = &As[ar1 * BK + ac1];
    unsigned short* BsW  = &Bs[br  * BK + bc];

    for (int k0 = 0; k0 < K; k0 += BK) {
        uint4 a0 = *(const uint4*)(Ag0 + k0);
        uint4 a1 = *(const uint4*)(Ag0 + k0 + 8);
        uint4 a2 = *(const uint4*)(Ag1 + k0);
        uint4 a3 = *(const uint4*)(Ag1 + k0 + 8);
        uint4 b0 = *(const uint4*)(Bg  + k0);
        uint4 b1 = *(const uint4*)(Bg  + k0 + 8);
        __syncthreads();
        *(uint4*)AsW0 = a0; *(uint4*)(AsW0 + 8) = a1;
        *(uint4*)AsW1 = a2; *(uint4*)(AsW1 + 8) = a3;
        *(uint4*)BsW  = b0; *(uint4*)(BsW  + 8) = b1;
        __syncthreads();

        Frag fa[4], fb[4];
#pragma unroll
        for (int m = 0; m < 4; ++m) {
            // A 16x32 bf16 layout: lanes 0-15 K{0..7,16..23}, lanes 16-31 K{8..15,24..31}
            int r = wrow * 64 + m * 16 + l15;
            fa[m].q[0] = *(const uint4*)&As[r * BK + 8 * hi];
            fa[m].q[1] = *(const uint4*)&As[r * BK + 16 + 8 * hi];
        }
#pragma unroll
        for (int n = 0; n < 4; ++n) {
            // B 32x16 bf16 layout: lanes 0-15 K{0..7,16..23}, lanes 16-31 K{8..15,24..31} (col = l15)
            int c = wcol * 64 + n * 16 + l15;
            fb[n].q[0] = *(const uint4*)&Bs[c * BK + 8 * hi];
            fb[n].q[1] = *(const uint4*)&Bs[c * BK + 16 + 8 * hi];
        }
#pragma unroll
        for (int m = 0; m < 4; ++m)
#pragma unroll
            for (int n = 0; n < 4; ++n)
                acc[m][n] = __builtin_amdgcn_wmma_f32_16x16x32_bf16(
                    false, fa[m].v, false, fb[n].v, (short)0, acc[m][n], false, false);
    }

    // Epilogue: C/D layout — VGPR r: lanes 0-15 M=r, lanes 16-31 M=8+r; N=l15
#pragma unroll
    for (int m = 0; m < 4; ++m) {
#pragma unroll
        for (int n = 0; n < 4; ++n) {
            int col = col0 + wcol * 64 + n * 16 + l15;
            float bv = bias[col];
            int rbase = row0 + wrow * 64 + m * 16 + hi * 8;
#pragma unroll
            for (int r = 0; r < 8; ++r) {
                float v = acc[m][n][r] + bv;
                size_t off = (size_t)(rbase + r) * N + col;
                if (gelu_bf16_out)
                    ((unsigned short*)Cout)[off] = f2bf(gelu_exact(v));
                else
                    ((float*)Cout)[off] = v;
            }
        }
    }
}

// ---------------------------------------------------------------------------
// Final routed mix: out = base*wm + adaptive*(1-wm)   (in place on d_out)
// ---------------------------------------------------------------------------
__global__ __launch_bounds__(256) void combine_kernel(
    float* __restrict__ out, const float* __restrict__ wmul, const int* __restrict__ idx,
    const float* __restrict__ xn, const float* __restrict__ ad0, const float* __restrict__ ad1)
{
    size_t i = (size_t)blockIdx.x * 256 + threadIdx.x;
    size_t e = i * 4;
    size_t token = e >> 10;                 // H_DIM == 1024
    float w  = wmul[token];
    int   id = idx[token];
    const float* selp = (id == 0) ? ad0 : ((id == 1) ? ad1 : xn);
    float4 bse = *(const float4*)(out + e);
    float4 sel = *(const float4*)(selp + e);
    float iw = 1.0f - w;
    float4 o;
    o.x = bse.x * w + sel.x * iw;
    o.y = bse.y * w + sel.y * iw;
    o.z = bse.z * w + sel.z * iw;
    o.w = bse.w * w + sel.w * iw;
    *(float4*)(out + e) = o;
}

// ---------------------------------------------------------------------------
extern "C" void kernel_launch(void* const* d_in, const int* in_sizes, int n_in,
                              void* d_out, int out_size, void* d_ws, size_t ws_size,
                              hipStream_t stream)
{
    const float* x        = (const float*)d_in[0];
    const float* wmul     = (const float*)d_in[1];
    const int*   widx     = (const int*)  d_in[2];
    const float* ln_in_g  = (const float*)d_in[3];
    const float* ln_in_b  = (const float*)d_in[4];
    const float* W1       = (const float*)d_in[5];
    const float* b1       = (const float*)d_in[6];
    const float* ln_h_g   = (const float*)d_in[7];
    const float* ln_h_b   = (const float*)d_in[8];
    const float* W2       = (const float*)d_in[9];
    const float* b2       = (const float*)d_in[10];
    const float* a1w1     = (const float*)d_in[11];
    const float* a1b1     = (const float*)d_in[12];
    const float* a1w2     = (const float*)d_in[13];
    const float* a1b2     = (const float*)d_in[14];
    const float* a2w1     = (const float*)d_in[15];
    const float* a2b1     = (const float*)d_in[16];
    const float* a2w2     = (const float*)d_in[17];
    const float* a2b2     = (const float*)d_in[18];

    char* ws = (char*)d_ws;
    size_t off = 0;
    auto carve = [&](size_t bytes) -> void* {
        void* p = ws + off;
        off += (bytes + 255) & ~(size_t)255;
        return p;
    };
    float*          xn_f  = (float*)         carve((size_t)NTOK * H_DIM * 4);
    unsigned short* xn_b  = (unsigned short*)carve((size_t)NTOK * H_DIM * 2);
    unsigned short* W1t   = (unsigned short*)carve((size_t)F_DIM * H_DIM * 2);
    unsigned short* W2t   = (unsigned short*)carve((size_t)H_DIM * F_DIM * 2);
    unsigned short* a1w1t = (unsigned short*)carve((size_t)256 * H_DIM * 2);
    unsigned short* a1w2t = (unsigned short*)carve((size_t)H_DIM * 256 * 2);
    unsigned short* a2w1t = (unsigned short*)carve((size_t)512 * H_DIM * 2);
    unsigned short* a2w2t = (unsigned short*)carve((size_t)H_DIM * 512 * 2);
    unsigned short* hpre  = (unsigned short*)carve((size_t)NTOK * F_DIM * 2);
    unsigned short* hnorm = (unsigned short*)carve((size_t)NTOK * F_DIM * 2);
    unsigned short* u0    = (unsigned short*)carve((size_t)NTOK * 256 * 2);
    unsigned short* u1    = (unsigned short*)carve((size_t)NTOK * 512 * 2);
    float*          ad0   = (float*)         carve((size_t)NTOK * H_DIM * 4);
    float*          ad1   = (float*)         carve((size_t)NTOK * H_DIM * 4);

    dim3 tb(32, 8);
    // W[K x F] -> Wt[F x K] bf16
    wt_transpose_kernel<<<dim3(F_DIM / 32, H_DIM / 32), tb, 0, stream>>>(W1, W1t, H_DIM, F_DIM);
    wt_transpose_kernel<<<dim3(H_DIM / 32, F_DIM / 32), tb, 0, stream>>>(W2, W2t, F_DIM, H_DIM);
    wt_transpose_kernel<<<dim3(256 / 32, H_DIM / 32), tb, 0, stream>>>(a1w1, a1w1t, H_DIM, 256);
    wt_transpose_kernel<<<dim3(H_DIM / 32, 256 / 32), tb, 0, stream>>>(a1w2, a1w2t, 256, H_DIM);
    wt_transpose_kernel<<<dim3(512 / 32, H_DIM / 32), tb, 0, stream>>>(a2w1, a2w1t, H_DIM, 512);
    wt_transpose_kernel<<<dim3(H_DIM / 32, 512 / 32), tb, 0, stream>>>(a2w2, a2w2t, 512, H_DIM);

    ln_x_kernel<<<NTOK, 256, 0, stream>>>(x, ln_in_g, ln_in_b, xn_f, xn_b);

    // base path
    gemm_bf16_kernel<<<dim3(F_DIM / BN, NTOK / BM), 256, 0, stream>>>(
        xn_b, W1t, b1, hpre, NTOK, F_DIM, H_DIM, 1);
    ln_h_kernel<<<NTOK, 256, 0, stream>>>(hpre, ln_h_g, ln_h_b, hnorm);
    gemm_bf16_kernel<<<dim3(H_DIM / BN, NTOK / BM), 256, 0, stream>>>(
        hnorm, W2t, b2, d_out, NTOK, H_DIM, F_DIM, 0);

    // adapter 256
    gemm_bf16_kernel<<<dim3(256 / BN, NTOK / BM), 256, 0, stream>>>(
        xn_b, a1w1t, a1b1, u0, NTOK, 256, H_DIM, 1);
    gemm_bf16_kernel<<<dim3(H_DIM / BN, NTOK / BM), 256, 0, stream>>>(
        u0, a1w2t, a1b2, ad0, NTOK, H_DIM, 256, 0);

    // adapter 512
    gemm_bf16_kernel<<<dim3(512 / BN, NTOK / BM), 256, 0, stream>>>(
        xn_b, a2w1t, a2b1, u1, NTOK, 512, H_DIM, 1);
    gemm_bf16_kernel<<<dim3(H_DIM / BN, NTOK / BM), 256, 0, stream>>>(
        u1, a2w2t, a2b2, ad1, NTOK, H_DIM, 512, 0);

    combine_kernel<<<(NTOK * H_DIM) / 1024, 256, 0, stream>>>(
        (float*)d_out, wmul, widx, xn_f, ad0, ad1);
}
